// TBlock_75711683493976
// MI455X (gfx1250) — matrix operand order repaired and checked by reading, not compiled, not run on hardware
//
#include <hip/hip_runtime.h>
#include <stdint.h>

#define D_MODEL 1024
#define NHEADS  16
#define HEADW   64
#define FFDIM   4096
#define BATCH   2
#define SEQ     2048
#define NTOK    (BATCH * SEQ)   // 4096

typedef __attribute__((ext_vector_type(16))) __bf16 v16bf;
typedef __attribute__((ext_vector_type(8)))  float  v8f;

struct BF16FragBits { uint4 lo, hi; };

static __device__ __forceinline__ v16bf frag_from(uint4 lo, uint4 hi) {
  BF16FragBits t{lo, hi};
  return __builtin_bit_cast(v16bf, t);
}

static __device__ __forceinline__ v8f wmma_bf16(v16bf a, v16bf b, v8f c) {
  // D = A(16x32) * B(32x16) + C, fp32 accumulate
  return __builtin_amdgcn_wmma_f32_16x16x32_bf16(false, a, false, b, (short)0, c, false, false);
}

static __device__ __forceinline__ unsigned short f2bf(float f) {
  unsigned int u = __float_as_uint(f);
  u += 0x7FFFu + ((u >> 16) & 1u);   // round-to-nearest-even
  return (unsigned short)(u >> 16);
}

// ---------------------------------------------------------------- weights f32 -> bf16
__global__ __launch_bounds__(256) void f32_to_bf16_kernel(const float* __restrict__ in,
                                                          unsigned short* __restrict__ out,
                                                          int n) {
  for (int i = blockIdx.x * blockDim.x + threadIdx.x; i < n; i += gridDim.x * blockDim.x)
    out[i] = f2bf(in[i]);
}

// ---------------------------------------------------------------- LayerNorm (one row / block)
__global__ __launch_bounds__(256) void layernorm_kernel(const float* __restrict__ x,
                                                        const float* __restrict__ g,
                                                        const float* __restrict__ beta,
                                                        float* __restrict__ out_f32,
                                                        unsigned short* __restrict__ out_bf) {
  const int row = blockIdx.x;
  const int t   = threadIdx.x;                       // 256 threads, one float4 each (D=1024)
  const float4* xr = reinterpret_cast<const float4*>(x) + (size_t)row * (D_MODEL / 4);
  float4 v = xr[t];
  float s  = v.x + v.y + v.z + v.w;
  float ss = v.x * v.x + v.y * v.y + v.z * v.z + v.w * v.w;
#pragma unroll
  for (int off = 16; off >= 1; off >>= 1) { s += __shfl_xor(s, off); ss += __shfl_xor(ss, off); }
  __shared__ float shs[8], shss[8];
  const int lane = t & 31, w = t >> 5;
  if (lane == 0) { shs[w] = s; shss[w] = ss; }
  __syncthreads();
  if (t < 32) {
    s  = (t < 8) ? shs[t]  : 0.f;
    ss = (t < 8) ? shss[t] : 0.f;
#pragma unroll
    for (int off = 4; off >= 1; off >>= 1) { s += __shfl_xor(s, off); ss += __shfl_xor(ss, off); }
    if (t == 0) { shs[0] = s; shss[0] = ss; }
  }
  __syncthreads();
  const float mu   = shs[0] * (1.f / D_MODEL);
  const float var  = shss[0] * (1.f / D_MODEL) - mu * mu;
  const float rstd = rsqrtf(var + 1e-5f);
  const float4 gv = reinterpret_cast<const float4*>(g)[t];
  const float4 bv = reinterpret_cast<const float4*>(beta)[t];
  float4 o;
  o.x = (v.x - mu) * rstd * gv.x + bv.x;
  o.y = (v.y - mu) * rstd * gv.y + bv.y;
  o.z = (v.z - mu) * rstd * gv.z + bv.z;
  o.w = (v.w - mu) * rstd * gv.w + bv.w;
  if (out_f32)
    reinterpret_cast<float4*>(out_f32)[(size_t)row * (D_MODEL / 4) + t] = o;
  if (out_bf)
    reinterpret_cast<ushort4*>(out_bf)[(size_t)row * (D_MODEL / 4) + t] =
        make_ushort4(f2bf(o.x), f2bf(o.y), f2bf(o.z), f2bf(o.w));
}

// ---------------------------------------------------------------- WMMA GEMM: C = A[M,K] * W[N,K]^T
// Block = 8 waves, block tile 128(M) x 128(N); wave tile 32(M) x 64(N):
// per 32-K chunk: 2 A-frags + 4 B-frags feed 8 WMMAs (each B reused twice).
enum { EPI_BF16 = 0, EPI_BF16_VT = 1, EPI_SILU_BF16 = 2, EPI_ADD_F32 = 3 };

template <int MODE>
__global__ __launch_bounds__(256) void gemm_bf16_wmma_kernel(
    const unsigned short* __restrict__ A,   // [M,K] bf16
    const unsigned short* __restrict__ W,   // [N,K] bf16
    void* __restrict__ OUTP,
    const float* __restrict__ RES,          // residual for EPI_ADD_F32
    int M, int N, int K) {
  const int lane  = threadIdx.x & 31;
  const int wave  = threadIdx.x >> 5;               // 8 waves / block
  const int l16   = lane & 15;
  const int lhalf = lane >> 4;
  const int m0 = blockIdx.y * 128 + (wave & 3) * 32;   // 4 wave-rows of 32
  const int n0 = blockIdx.x * 128 + (wave >> 2) * 64;  // 2 wave-cols of 64

  const v8f zero = {0.f, 0.f, 0.f, 0.f, 0.f, 0.f, 0.f, 0.f};
  v8f acc[2][4] = {{zero, zero, zero, zero}, {zero, zero, zero, zero}};

  const unsigned short* arow[2];
#pragma unroll
  for (int mi = 0; mi < 2; ++mi)
    arow[mi] = A + (size_t)(m0 + 16 * mi + l16) * K + 16 * lhalf;
  const unsigned short* wrow[4];
#pragma unroll
  for (int f = 0; f < 4; ++f)
    wrow[f] = W + (size_t)(n0 + 16 * f + l16) * K + 16 * lhalf;

  for (int k0 = 0; k0 < K; k0 += 32) {
    if (k0 + 256 < K) {                    // stream prefetch (global_prefetch_b8)
      __builtin_prefetch(arow[0] + k0 + 256, 0, 3);
      __builtin_prefetch(wrow[0] + k0 + 256, 0, 3);
    }
    // A fragments, 16x32 bf16: 16 contiguous values at k0+16*lhalf
    v16bf af[2];
#pragma unroll
    for (int mi = 0; mi < 2; ++mi) {
      const unsigned short* pa = arow[mi] + k0;
      af[mi] = frag_from(*(const uint4*)pa, *(const uint4*)(pa + 8));
    }
#pragma unroll
    for (int f = 0; f < 4; ++f) {
      // B fragment, 32x16 bf16: 16 contiguous values at k0+16*lhalf of weight row n
      const unsigned short* pb = wrow[f] + k0;
      v16bf bfrag = frag_from(*(const uint4*)pb, *(const uint4*)(pb + 8));
      acc[0][f] = wmma_bf16(af[0], bfrag, acc[0][f]);
      acc[1][f] = wmma_bf16(af[1], bfrag, acc[1][f]);
    }
  }

#pragma unroll
  for (int mi = 0; mi < 2; ++mi) {
#pragma unroll
    for (int f = 0; f < 4; ++f) {
#pragma unroll
      for (int r = 0; r < 8; ++r) {
        const int row = m0 + 16 * mi + r + 8 * lhalf;  // C/D layout: VGPR r -> M=r (+8 hi lanes)
        const int col = n0 + 16 * f + l16;
        const float v = acc[mi][f][r];
        if (MODE == EPI_BF16) {
          ((unsigned short*)OUTP)[(size_t)row * N + col] = f2bf(v);
        } else if (MODE == EPI_BF16_VT) {  // store V transposed per head: [B,H,HW,T]
          const int bb = row >> 11, tt = row & (SEQ - 1);
          const int hh = col >> 6,  dd = col & 63;
          ((unsigned short*)OUTP)[((size_t)(bb * NHEADS + hh) * HEADW + dd) * SEQ + tt] = f2bf(v);
        } else if (MODE == EPI_SILU_BF16) {
          const float sv = v / (1.f + __expf(-v));
          ((unsigned short*)OUTP)[(size_t)row * N + col] = f2bf(sv);
        } else {                           // EPI_ADD_F32: out = residual + acc
          const size_t idx = (size_t)row * N + col;
          ((float*)OUTP)[idx] = RES[idx] + v;
        }
      }
    }
  }
}

// ---------------------------------------------------------------- Flash attention (ALiBi + causal)
__global__ __launch_bounds__(256) void attn_kernel(const unsigned short* __restrict__ q,
                                                   const unsigned short* __restrict__ k,
                                                   const unsigned short* __restrict__ vT,
                                                   const float* __restrict__ xn,
                                                   float* __restrict__ att_out) {
  const int lane  = threadIdx.x & 31;
  const int wave  = threadIdx.x >> 5;
  const int l16   = lane & 15;
  const int lhalf = lane >> 4;
  const int bh = blockIdx.x;            // B*H = 32
  const int b  = bh >> 4;
  const int h  = bh & 15;
  const int q0 = (blockIdx.y * 8 + wave) * 16;   // this wave's 16 query rows

  __shared__ __align__(16) unsigned short Pst[8][16 * 32];  // wave-private P staging
  unsigned short* Pw = Pst[wave];

  // Q fragments (16x64 -> two 16x32 A-frags), contiguous in head dim
  const unsigned short* qrow = q + (size_t)(b * SEQ + q0 + l16) * D_MODEL + h * HEADW;
  v16bf qa[2];
#pragma unroll
  for (int c = 0; c < 2; ++c) {
    const unsigned short* p = qrow + c * 32 + 16 * lhalf;
    qa[c] = frag_from(*(const uint4*)p, *(const uint4*)(p + 8));
  }

  const v8f zero = {0.f, 0.f, 0.f, 0.f, 0.f, 0.f, 0.f, 0.f};
  v8f o[4] = {zero, zero, zero, zero};
  float m_run[8], l_run[8];
  const float slope = exp2f(-0.5f * (float)(h + 1));   // 2^(-(h+1)*8/H), H=16

  // Hoisted per-row ALiBi constants: slope*(col-row) = slope*j0 + ab{0,1}[r];
  // causal mask (col>row) <=> j0 > thr{0,1}[r].
  float ab0[8], ab1[8];
  int thr0[8], thr1[8];
#pragma unroll
  for (int r = 0; r < 8; ++r) {
    const int row = q0 + r + 8 * lhalf;
    ab0[r] = slope * (float)(l16 - row);
    ab1[r] = slope * (float)(l16 + 16 - row);
    thr0[r] = row - l16;
    thr1[r] = row - l16 - 16;
    m_run[r] = -1e30f;
    l_run[r] = 0.f;
  }

  for (int j0 = 0; j0 < q0 + 16; j0 += 32) {           // causal kv range, chunks of 32
    // ---- scores S = Q K^T for two 16-key tiles
    v8f s[2] = {zero, zero};
#pragma unroll
    for (int tt = 0; tt < 2; ++tt) {
      const unsigned short* krow =
          k + (size_t)(b * SEQ + j0 + 16 * tt + l16) * D_MODEL + h * HEADW + 16 * lhalf;
#pragma unroll
      for (int c = 0; c < 2; ++c) {
        const unsigned short* p = krow + c * 32;
        v16bf kb = frag_from(*(const uint4*)p, *(const uint4*)(p + 8));
        s[tt] = wmma_bf16(qa[c], kb, s[tt]);
      }
    }
    // ---- online softmax: scale + alibi + causal mask, row max/sum via 16-lane shuffles
    const float sj = slope * (float)j0;
    float alpha[8];
#pragma unroll
    for (int r = 0; r < 8; ++r) {
      float v0 = fmaf(s[0][r], 0.125f, sj + ab0[r]);
      float v1 = fmaf(s[1][r], 0.125f, sj + ab1[r]);
      if (j0 > thr0[r]) v0 = -1e30f;
      if (j0 > thr1[r]) v1 = -1e30f;
      float rm = fmaxf(v0, v1);
#pragma unroll
      for (int off = 8; off >= 1; off >>= 1) rm = fmaxf(rm, __shfl_xor(rm, off));
      const float mnew = fmaxf(m_run[r], rm);
      alpha[r] = __expf(m_run[r] - mnew);
      const float p0 = __expf(v0 - mnew);
      const float p1 = __expf(v1 - mnew);
      Pw[(r + 8 * lhalf) * 32 + l16]      = f2bf(p0);   // C-layout -> row-major P in LDS
      Pw[(r + 8 * lhalf) * 32 + 16 + l16] = f2bf(p1);
      float ps = p0 + p1;
#pragma unroll
      for (int off = 8; off >= 1; off >>= 1) ps += __shfl_xor(ps, off);
      l_run[r] = l_run[r] * alpha[r] + ps;
      m_run[r] = mnew;
    }
#pragma unroll
    for (int f = 0; f < 4; ++f)
#pragma unroll
      for (int r = 0; r < 8; ++r) o[f][r] *= alpha[r];

    // LDS ops are in-order per wave; wait + compiler barrier before re-reading P
    asm volatile("s_wait_dscnt 0" ::: "memory");

    // P as A-fragment (16x32)
    const unsigned short* pp = Pw + l16 * 32 + 16 * lhalf;
    v16bf pa = frag_from(*(const uint4*)pp, *(const uint4*)(pp + 8));

    // O += P * V, V pre-transposed to [B,H,HW,T] so B-frags are contiguous in T
#pragma unroll
    for (int f = 0; f < 4; ++f) {
      const unsigned short* vp =
          vT + ((size_t)(b * NHEADS + h) * HEADW + f * 16 + l16) * SEQ + j0 + 16 * lhalf;
      v16bf vb = frag_from(*(const uint4*)vp, *(const uint4*)(vp + 8));
      o[f] = wmma_bf16(pa, vb, o[f]);
    }
  }

  // ---- epilogue: att_out = xn + softmax(..)V
#pragma unroll
  for (int r = 0; r < 8; ++r) {
    const float inv = 1.f / l_run[r];
    const int row = q0 + r + 8 * lhalf;
#pragma unroll
    for (int f = 0; f < 4; ++f) {
      const size_t idx = (size_t)(b * SEQ + row) * D_MODEL + h * HEADW + f * 16 + l16;
      att_out[idx] = xn[idx] + o[f][r] * inv;
    }
  }
}

// ---------------------------------------------------------------- host-side launch
extern "C" void kernel_launch(void* const* d_in, const int* in_sizes, int n_in,
                              void* d_out, int out_size, void* d_ws, size_t ws_size,
                              hipStream_t stream) {
  const float* x  = (const float*)d_in[0];
  const float* Wq = (const float*)d_in[1];
  const float* Wk = (const float*)d_in[2];
  const float* Wv = (const float*)d_in[3];
  const float* W1 = (const float*)d_in[4];
  const float* W2 = (const float*)d_in[5];
  const float* g1 = (const float*)d_in[6];
  const float* b1 = (const float*)d_in[7];
  const float* g2 = (const float*)d_in[8];
  const float* b2 = (const float*)d_in[9];
  float* out = (float*)d_out;

  char* p = (char*)d_ws;
  auto carve = [&](size_t bytes) -> char* {
    char* r = p;
    p += (bytes + 255) & ~(size_t)255;
    return r;
  };
  float*          xn_f32  = (float*)         carve((size_t)NTOK * D_MODEL * 4);
  unsigned short* xn_bf   = (unsigned short*)carve((size_t)NTOK * D_MODEL * 2);
  unsigned short* q_bf    = (unsigned short*)carve((size_t)NTOK * D_MODEL * 2);
  unsigned short* k_bf    = (unsigned short*)carve((size_t)NTOK * D_MODEL * 2);
  unsigned short* vT_bf   = (unsigned short*)carve((size_t)NTOK * D_MODEL * 2);
  float*          att_out = (float*)         carve((size_t)NTOK * D_MODEL * 4);
  unsigned short* h_bf    = (unsigned short*)carve((size_t)NTOK * D_MODEL * 2);
  unsigned short* m1_bf   = (unsigned short*)carve((size_t)NTOK * FFDIM * 2);
  unsigned short* Wq_bf   = (unsigned short*)carve((size_t)D_MODEL * D_MODEL * 2);
  unsigned short* Wk_bf   = (unsigned short*)carve((size_t)D_MODEL * D_MODEL * 2);
  unsigned short* Wv_bf   = (unsigned short*)carve((size_t)D_MODEL * D_MODEL * 2);
  unsigned short* W1_bf   = (unsigned short*)carve((size_t)FFDIM * D_MODEL * 2);
  unsigned short* W2_bf   = (unsigned short*)carve((size_t)D_MODEL * FFDIM * 2);

  const dim3 blk(256);

  // weights -> bf16
  f32_to_bf16_kernel<<<512, blk, 0, stream>>>(Wq, Wq_bf, D_MODEL * D_MODEL);
  f32_to_bf16_kernel<<<512, blk, 0, stream>>>(Wk, Wk_bf, D_MODEL * D_MODEL);
  f32_to_bf16_kernel<<<512, blk, 0, stream>>>(Wv, Wv_bf, D_MODEL * D_MODEL);
  f32_to_bf16_kernel<<<512, blk, 0, stream>>>(W1, W1_bf, FFDIM * D_MODEL);
  f32_to_bf16_kernel<<<512, blk, 0, stream>>>(W2, W2_bf, D_MODEL * FFDIM);

  // LN1 -> xn (f32 for residual, bf16 for GEMMs)
  layernorm_kernel<<<NTOK, blk, 0, stream>>>(x, g1, b1, xn_f32, xn_bf);

  // QKV projections
  gemm_bf16_wmma_kernel<EPI_BF16><<<dim3(D_MODEL / 128, NTOK / 128), blk, 0, stream>>>(
      xn_bf, Wq_bf, q_bf, nullptr, NTOK, D_MODEL, D_MODEL);
  gemm_bf16_wmma_kernel<EPI_BF16><<<dim3(D_MODEL / 128, NTOK / 128), blk, 0, stream>>>(
      xn_bf, Wk_bf, k_bf, nullptr, NTOK, D_MODEL, D_MODEL);
  gemm_bf16_wmma_kernel<EPI_BF16_VT><<<dim3(D_MODEL / 128, NTOK / 128), blk, 0, stream>>>(
      xn_bf, Wv_bf, vT_bf, nullptr, NTOK, D_MODEL, D_MODEL);

  // attention + residual (att_out = xn + O)
  attn_kernel<<<dim3(BATCH * NHEADS, SEQ / 128), blk, 0, stream>>>(
      q_bf, k_bf, vT_bf, xn_f32, att_out);

  // LN2 -> h (bf16 only)
  layernorm_kernel<<<NTOK, blk, 0, stream>>>(att_out, g2, b2, nullptr, h_bf);

  // MLP: m1 = silu(h @ W1^T), out = att_out + m1 @ W2^T
  gemm_bf16_wmma_kernel<EPI_SILU_BF16><<<dim3(FFDIM / 128, NTOK / 128), blk, 0, stream>>>(
      h_bf, W1_bf, m1_bf, nullptr, NTOK, FFDIM, D_MODEL);
  gemm_bf16_wmma_kernel<EPI_ADD_F32><<<dim3(D_MODEL / 128, NTOK / 128), blk, 0, stream>>>(
      m1_bf, W2_bf, out, att_out, NTOK, D_MODEL, FFDIM);
}
